// AuxiliaryConditionerBlock_33380485825343
// MI455X (gfx1250) — compile-verified, matrix operand order checked by reading, not run on hardware
//
#include <hip/hip_runtime.h>
#include <hip/hip_bf16.h>
#include <cstdint>

// ---- constants for this problem ----
#define SEQ   2048
#define DDIM  256
#define NH3   48          // 3*H heads total
#define CDIM  64
#define KQ_LD (NH3 * CDIM)   // 3072 elements per seq row in Kb/Qb
#define VT_LD SEQ            // VbT is (48*256, 2048)

typedef __attribute__((ext_vector_type(16))) __bf16 v16bf;
typedef __attribute__((ext_vector_type(8)))  __bf16 bf16x8;
typedef __attribute__((ext_vector_type(8)))  float  v8f;

union FragU { struct { bf16x8 lo, hi; } p; v16bf v; };

// A/B fragment gather for 16-bit 16x16x32 WMMA from a row-major source:
// caller passes pointer to (row = lane%16) with contraction dim contiguous.
__device__ __forceinline__ v16bf load_frag(const __bf16* rowptr, int half) {
  FragU u;
  u.p.lo = *reinterpret_cast<const bf16x8*>(rowptr + 8 * half);
  u.p.hi = *reinterpret_cast<const bf16x8*>(rowptr + 16 + 8 * half);
  return u.v;
}

__device__ __forceinline__ __bf16 f2bf(float f) {
  union { float f; unsigned int u; } v; v.f = f;
  unsigned int r = v.u + 0x7FFFu + ((v.u >> 16) & 1u);   // round-to-nearest-even
  unsigned short h = (unsigned short)(r >> 16);
  __bf16 b;
  __builtin_memcpy(&b, &h, 2);
  return b;
}

// ---------------- prep kernels ----------------
__global__ void f32_to_bf16_k(const float* __restrict__ src, __bf16* __restrict__ dst, int n) {
  int i = blockIdx.x * blockDim.x + threadIdx.x;
  if (i < n) dst[i] = f2bf(src[i]);
}

__global__ void zero_f32_k(float* __restrict__ p, int n) {
  int i = blockIdx.x * blockDim.x + threadIdx.x;
  if (i < n) p[i] = 0.0f;
}

// pos (dim 6) and rot (dim 4) projections -> heads 16..47 of Kb/Qb. Tiny FLOPs -> VALU.
// K values are pre-scaled: pos heads x0.25 (1/sqrt(H)); rot heads x0.5 (squared later).
__global__ void posrot_kq_k(const float* __restrict__ pos, const float* __restrict__ rotq,
                            const float* __restrict__ Wp, const float* __restrict__ bp,
                            const float* __restrict__ Wr,
                            __bf16* __restrict__ Kb, __bf16* __restrict__ Qb) {
  int gid = blockIdx.x * blockDim.x + threadIdx.x;    // SEQ * 4096 threads
  int s   = gid >> 12;
  int idx = gid & 4095;
  int hl  = idx >> 7;        // 0..31
  int rr  = idx & 127;       // 0..127 (k | q)
  float val;
  float kscale;
  if (hl < 16) {             // pos-source heads
    const float* w = Wp + (hl * 128 + rr) * 6;
    const float* x = pos + s * 6;
    val = bp[hl * 128 + rr];
#pragma unroll
    for (int t = 0; t < 6; ++t) val += x[t] * w[t];
    kscale = 0.25f;
  } else {                   // rot-source heads (no bias)
    int h2 = hl - 16;
    const float* w = Wr + (h2 * 128 + rr) * 4;
    const float* x = rotq + s * 4;
    val = 0.0f;
#pragma unroll
    for (int t = 0; t < 4; ++t) val += x[t] * w[t];
    kscale = 0.5f;
  }
  int h = 16 + hl;
  if (rr < 64) Kb[s * KQ_LD + h * CDIM + rr]        = f2bf(val * kscale);
  else         Qb[s * KQ_LD + h * CDIM + (rr - 64)] = f2bf(val);
}

// ---------------- projection GEMM: Y = X(2048x256) @ W^T(N x 256) + bias ----------------
// one wave computes a 16x64 tile; mode 0: store transposed into VbT[n*2048+s]
// mode 1 (N=2048): scatter into Kb/Qb for heads 0..15 (K pre-scaled by 0.25)
__global__ __launch_bounds__(128)
void gemm_proj_k(const __bf16* __restrict__ X, const __bf16* __restrict__ W,
                 const float* __restrict__ bias, int N, int mode,
                 __bf16* __restrict__ out0, __bf16* __restrict__ out1) {
  int wave = blockIdx.x * 4 + (threadIdx.x >> 5);
  int lane = threadIdx.x & 31;
  int l16  = lane & 15;
  int half = lane >> 4;
  int ntiles = N >> 6;
  int i0 = (wave / ntiles) << 4;
  int n0 = (wave % ntiles) << 6;

  v8f zero = {};
  v8f acc[4];
#pragma unroll
  for (int f = 0; f < 4; ++f) acc[f] = zero;

  const __bf16* xrow = X + (i0 + l16) * 256;
#pragma unroll
  for (int k0 = 0; k0 < 256; k0 += 32) {
    v16bf a = load_frag(xrow + k0, half);
#pragma unroll
    for (int f = 0; f < 4; ++f) {
      v16bf b = load_frag(W + (n0 + f * 16 + l16) * 256 + k0, half);
      acc[f] = __builtin_amdgcn_wmma_f32_16x16x32_bf16(false, a, false, b,
                                                       (short)0, acc[f], false, false);
    }
  }
#pragma unroll
  for (int f = 0; f < 4; ++f) {
    int n = n0 + f * 16 + l16;
    float bb = bias[n];
#pragma unroll
    for (int r = 0; r < 8; ++r) {
      int s = i0 + r + 8 * half;
      float y = acc[f][r] + bb;
      if (mode == 0) {
        out0[n * VT_LD + s] = f2bf(y);                 // VbT transposed store
      } else {
        int h  = n >> 7;
        int rr = n & 127;
        if (rr < 64) out0[s * KQ_LD + h * CDIM + rr]        = f2bf(y * 0.25f); // Kb
        else         out1[s * KQ_LD + h * CDIM + (rr - 64)] = f2bf(y);         // Qb
      }
    }
  }
}

// ---------------- fused flash attention ----------------
// one wave = (head h, 16 query rows). Streams j in chunks of 32.
// K already carries the 1/sqrt(H) scaling (as sqrt for rot heads).
__global__ __launch_bounds__(128)
void attn_k(const __bf16* __restrict__ Kb, const __bf16* __restrict__ Qb,
            const __bf16* __restrict__ VbT, float* __restrict__ out) {
  __shared__ __align__(16) __bf16 plds[4][16 * 32];
  int warp = threadIdx.x >> 5;
  int lane = threadIdx.x & 31;
  int l16  = lane & 15;
  int half = lane >> 4;
  int task = blockIdx.x * 4 + warp;       // 0..6143
  int h    = task >> 7;                   // 0..47
  int i0   = (task & 127) << 4;
  bool rot = (h >= 32);
  __bf16* pl = plds[warp];

  const __bf16* krow = Kb + (i0 + l16) * KQ_LD + h * CDIM;
  v16bf ka0 = load_frag(krow, half);
  v16bf ka1 = load_frag(krow + 32, half);

  v8f zero = {};
  v8f O[16];
#pragma unroll
  for (int t = 0; t < 16; ++t) O[t] = zero;
  float m_run[8], l_acc[8];
#pragma unroll
  for (int r = 0; r < 8; ++r) { m_run[r] = -1e30f; l_acc[r] = 0.0f; }

  for (int j0 = 0; j0 < SEQ; j0 += 32) {
    // ---- both 16x16 logit tiles of this 32-key chunk ----
    const __bf16* qrow0 = Qb + (j0 + l16) * KQ_LD + h * CDIM;
    const __bf16* qrow1 = qrow0 + 16 * KQ_LD;
    v8f c0 = zero, c1 = zero;
    c0 = __builtin_amdgcn_wmma_f32_16x16x32_bf16(false, ka0, false, load_frag(qrow0, half),      (short)0, c0, false, false);
    c0 = __builtin_amdgcn_wmma_f32_16x16x32_bf16(false, ka1, false, load_frag(qrow0 + 32, half), (short)0, c0, false, false);
    c1 = __builtin_amdgcn_wmma_f32_16x16x32_bf16(false, ka0, false, load_frag(qrow1, half),      (short)0, c1, false, false);
    c1 = __builtin_amdgcn_wmma_f32_16x16x32_bf16(false, ka1, false, load_frag(qrow1 + 32, half), (short)0, c1, false, false);
    if (rot) {                            // rotation heads: logits squared (scale folded in K)
#pragma unroll
      for (int r = 0; r < 8; ++r) { c0[r] = c0[r] * c0[r]; c1[r] = c1[r] * c1[r]; }
    }
    // ---- one softmax-stats update per chunk ----
    float sc[8];
    bool need = false;
#pragma unroll
    for (int r = 0; r < 8; ++r) {
      float mx = fmaxf(c0[r], c1[r]);     // row max over both tiles
      mx = fmaxf(mx, __shfl_xor(mx, 1, 32));
      mx = fmaxf(mx, __shfl_xor(mx, 2, 32));
      mx = fmaxf(mx, __shfl_xor(mx, 4, 32));
      mx = fmaxf(mx, __shfl_xor(mx, 8, 32));
      float mn = fmaxf(m_run[r], mx);
      sc[r] = __expf(m_run[r] - mn);      // == 1.0f exactly when max unchanged
      m_run[r] = mn;
      need |= (sc[r] != 1.0f);
    }
    if (__any(need)) {                    // rescale accumulators only when max moved
#pragma unroll
      for (int r = 0; r < 8; ++r) {
        l_acc[r] *= sc[r];
#pragma unroll
        for (int t = 0; t < 16; ++t) O[t][r] *= sc[r];
      }
    }
    // ---- exponentiate, accumulate per-lane partial denominator, stage P tile ----
#pragma unroll
    for (int r = 0; r < 8; ++r) {
      float p0 = __expf(c0[r] - m_run[r]);
      float p1 = __expf(c1[r] - m_run[r]);
      l_acc[r] += p0 + p1;                // lane-local column sums; reduced in epilogue
      pl[(r + 8 * half) * 32 + l16]      = f2bf(p0);
      pl[(r + 8 * half) * 32 + 16 + l16] = f2bf(p1);
    }
    asm volatile("s_wait_dscnt 0" ::: "memory");       // LDS transpose RAW fence
    v16bf pa = load_frag(pl + l16 * 32, half);
#pragma unroll
    for (int dt = 0; dt < 16; ++dt) {
      v16bf vb = load_frag(VbT + (h * 256 + dt * 16 + l16) * VT_LD + j0, half);
      O[dt] = __builtin_amdgcn_wmma_f32_16x16x32_bf16(false, pa, false, vb,
                                                      (short)0, O[dt], false, false);
    }
  }
  // ---- epilogue: finish denominator reduction, normalize, accumulate output ----
#pragma unroll
  for (int r = 0; r < 8; ++r) {
    float lsum = l_acc[r];
    lsum += __shfl_xor(lsum, 1, 32);
    lsum += __shfl_xor(lsum, 2, 32);
    lsum += __shfl_xor(lsum, 4, 32);
    lsum += __shfl_xor(lsum, 8, 32);
    float inv = 1.0f / lsum;
    int row = i0 + r + 8 * half;
#pragma unroll
    for (int dt = 0; dt < 16; ++dt)
      atomicAdd(out + row * DDIM + dt * 16 + l16, O[dt][r] * inv);
  }
}

extern "C" void kernel_launch(void* const* d_in, const int* in_sizes, int n_in,
                              void* d_out, int out_size, void* d_ws, size_t ws_size,
                              hipStream_t stream) {
  (void)in_sizes; (void)n_in; (void)out_size; (void)ws_size;
  const float* nodes = (const float*)d_in[0];
  const float* pos   = (const float*)d_in[1];
  const float* rotq  = (const float*)d_in[2];
  const float* Wn    = (const float*)d_in[3];
  const float* bn    = (const float*)d_in[4];
  const float* Wp    = (const float*)d_in[5];
  const float* bp    = (const float*)d_in[6];
  const float* Wr    = (const float*)d_in[7];
  const float* Wv    = (const float*)d_in[8];
  const float* bv    = (const float*)d_in[9];
  float* out = (float*)d_out;

  char* ws = (char*)d_ws;
  __bf16* nodes_bf = (__bf16*)ws; ws += (size_t)SEQ * 256 * 2;          // 1 MB
  __bf16* Wn_bf    = (__bf16*)ws; ws += (size_t)2048 * 256 * 2;         // 1 MB
  __bf16* Wv_bf    = (__bf16*)ws; ws += (size_t)12288 * 256 * 2;        // 6 MB
  __bf16* Kb       = (__bf16*)ws; ws += (size_t)SEQ * KQ_LD * 2;        // 12.6 MB
  __bf16* Qb       = (__bf16*)ws; ws += (size_t)SEQ * KQ_LD * 2;        // 12.6 MB
  __bf16* VbT      = (__bf16*)ws; ws += (size_t)12288 * SEQ * 2;        // 50.3 MB

  f32_to_bf16_k<<<(SEQ * 256 + 255) / 256, 256, 0, stream>>>(nodes, nodes_bf, SEQ * 256);
  f32_to_bf16_k<<<(2048 * 256 + 255) / 256, 256, 0, stream>>>(Wn, Wn_bf, 2048 * 256);
  f32_to_bf16_k<<<(12288 * 256 + 255) / 256, 256, 0, stream>>>(Wv, Wv_bf, 12288 * 256);
  zero_f32_k<<<(SEQ * DDIM + 255) / 256, 256, 0, stream>>>(out, SEQ * DDIM);
  posrot_kq_k<<<(SEQ * 4096) / 256, 256, 0, stream>>>(pos, rotq, Wp, bp, Wr, Kb, Qb);

  // K/Q from nodes (heads 0..15): waves = 128 * (2048/64) = 4096 -> 1024 blocks
  gemm_proj_k<<<1024, 128, 0, stream>>>(nodes_bf, Wn_bf, bn, 2048, 1, Kb, Qb);
  // V (transposed store): waves = 128 * (12288/64) = 24576 -> 6144 blocks
  gemm_proj_k<<<6144, 128, 0, stream>>>(nodes_bf, Wv_bf, bv, 12288, 0, VbT, nullptr);

  // attention: 6144 wave-tasks / 4 waves per block
  attn_k<<<1536, 128, 0, stream>>>(Kb, Qb, VbT, out);
}
